// PolyLoRALinear_89146341195908
// MI455X (gfx1250) — matrix operand order repaired and checked by reading, not compiled, not run on hardware
//
#include <hip/hip_runtime.h>
#include <hip/hip_bf16.h>

// ---------------------------------------------------------------------------
// PolyLoRALinear on MI455X (gfx1250, wave32, WMMA bf16 16x16x32)
// bs=8, seq=2048, d_in=d_out=4096, n_splits=1, n_skills=8, rank=16
// ---------------------------------------------------------------------------

#define BS      8
#define SEQ     2048
#define D_IN    4096
#define D_OUT   4096
#define NSKILL  8
#define RANK    16
#define M_TOTAL (BS * SEQ)   // 16384

typedef __attribute__((ext_vector_type(16))) __bf16 v16bf;
typedef __attribute__((ext_vector_type(8)))  float  v8f;

union FragB {
    v16bf    v;
    unsigned u[8];
    uint4    q[2];
};

static __device__ __forceinline__ unsigned fbits(float a) {
    union { float f; unsigned u; } x; x.f = a;
    return x.u;
}

// f32 pair -> packed bf16 pair, round-to-nearest (ties away from zero).
// 2 adds + 1 v_perm_b32 — far cheaper than the RNE bit-twiddle chain.
static __device__ __forceinline__ unsigned pk_bf16(float a, float b) {
    unsigned au = fbits(a) + 0x8000u;
    unsigned bu = fbits(b) + 0x8000u;
#if __has_builtin(__builtin_amdgcn_perm)
    // bytes 0-3 of {src0:src1} are src1, 4-7 are src0; pick hi16 of each
    return __builtin_amdgcn_perm(bu, au, 0x07060302u);
#else
    return (au >> 16) | (bu & 0xFFFF0000u);
#endif
}

static __device__ __forceinline__ unsigned short f2bf(float a) {
    return (unsigned short)((fbits(a) + 0x8000u) >> 16);
}

// sigmoid + sum-normalize over skills for one batch element
static __device__ __forceinline__ void compute_mw(const int* task_ids,
                                                  const float* module_logits,
                                                  int b, float* mw) {
    int t = task_ids[b];
    float s = 0.f;
#pragma unroll
    for (int i = 0; i < NSKILL; ++i) {
        float z = module_logits[t * NSKILL + i];
        float e = 1.f / (1.f + __expf(-z));
        mw[i] = e; s += e;
    }
    float inv = 1.f / (s + 1e-12f);
#pragma unroll
    for (int i = 0; i < NSKILL; ++i) mw[i] *= inv;
}

// ---------------------------------------------------------------------------
// Kernel 1a: AmixT[b][r][i] = sum_s mw[b][s] * lora_a[s][i][r]   (bf16)
// ---------------------------------------------------------------------------
__global__ __launch_bounds__(256) void poly_mix_a(
    const float* __restrict__ lora_a,        // [8][4096][16]
    const float* __restrict__ module_logits, // [256][8]
    const int*   __restrict__ task_ids,      // [8]
    unsigned short* __restrict__ amixT)      // [8][16][4096] bf16
{
    int idx = blockIdx.x * 256 + threadIdx.x;     // 0 .. 32767
    int b = idx >> 12;
    int i = idx & (D_IN - 1);
    float mw[NSKILL];
    compute_mw(task_ids, module_logits, b, mw);

    float acc[RANK];
#pragma unroll
    for (int r = 0; r < RANK; ++r) acc[r] = 0.f;

#pragma unroll
    for (int s = 0; s < NSKILL; ++s) {
        float w = mw[s];
        const float4* ap = (const float4*)(lora_a + ((size_t)(s * D_IN + i)) * RANK);
#pragma unroll
        for (int j = 0; j < 4; ++j) {
            float4 p = ap[j];
            acc[4 * j + 0] += w * p.x;
            acc[4 * j + 1] += w * p.y;
            acc[4 * j + 2] += w * p.z;
            acc[4 * j + 3] += w * p.w;
        }
    }
#pragma unroll
    for (int r = 0; r < RANK; ++r)
        amixT[(((size_t)b * RANK + r) << 12) + i] = f2bf(acc[r]);
}

// ---------------------------------------------------------------------------
// Kernel 1b: BmixT[b][n][r] = (1/rank) * sum_s mw[b][s] * lora_b[s][r][n]
// ---------------------------------------------------------------------------
__global__ __launch_bounds__(256) void poly_mix_b(
    const float* __restrict__ lora_b,        // [8][16][4096]
    const float* __restrict__ module_logits,
    const int*   __restrict__ task_ids,
    unsigned short* __restrict__ bmixT)      // [8][4096][16] bf16
{
    int idx = blockIdx.x * 256 + threadIdx.x;
    int b = idx >> 12;
    int n = idx & (D_OUT - 1);
    float mw[NSKILL];
    compute_mw(task_ids, module_logits, b, mw);

    float acc[RANK];
#pragma unroll
    for (int r = 0; r < RANK; ++r) acc[r] = 0.f;
#pragma unroll
    for (int s = 0; s < NSKILL; ++s) {
        float w = mw[s];
#pragma unroll
        for (int r = 0; r < RANK; ++r)
            acc[r] += w * lora_b[((size_t)(s * RANK + r) << 12) + n];
    }
    union { unsigned u[8]; uint4 q[2]; } pk;
#pragma unroll
    for (int j = 0; j < 8; ++j)
        pk.u[j] = pk_bf16(acc[2 * j] * (1.0f / RANK), acc[2 * j + 1] * (1.0f / RANK));
    uint4* dst = (uint4*)(bmixT + (((size_t)b << 12) + n) * RANK);
    dst[0] = pk.q[0];
    dst[1] = pk.q[1];
}

// ---------------------------------------------------------------------------
// Kernel 2: xa[m][r] = sum_i x[m][i] * Amix[b][i][r]   (WMMA, out bf16)
// block = 8 waves, each wave owns 16 rows x 16 ranks; K loop over d_in
// ---------------------------------------------------------------------------
__global__ __launch_bounds__(256) void poly_xa(
    const float* __restrict__ x,             // [16384][4096]
    const unsigned short* __restrict__ amixT,// [8][16][4096] bf16
    unsigned short* __restrict__ xa)         // [16384][16] bf16
{
    int wave = threadIdx.x >> 5;
    int lane = threadIdx.x & 31;
    int g = lane >> 4;          // K half
    int r = lane & 15;          // A row / B column / D column
    int rowBase = blockIdx.x * 128 + wave * 16;
    int b = rowBase >> 11;      // batch (2048 rows per batch)
    int arow = rowBase + r;

    v8f acc = {0.f, 0.f, 0.f, 0.f, 0.f, 0.f, 0.f, 0.f};
    const float* xp = x + (size_t)arow * D_IN;
    const unsigned short* bp = amixT + ((size_t)(b * RANK + r) << 12);

    for (int k0 = 0; k0 < D_IN; k0 += 32) {
        FragB A, B;
        // warm L2->WGP path one K-step ahead (global_prefetch_b8)
        if (k0 + 32 < D_IN)
            __builtin_prefetch(xp + k0 + 32 + g * 8, 0, 1);
        // A: 16x32 bf16 fragment built from f32 x (ISA 16-bit A layout)
        const float4* xr0 = (const float4*)(xp + k0 + g * 8);        // K = g*8 .. g*8+7
        const float4* xr1 = (const float4*)(xp + k0 + 16 + g * 8);   // K = 16+g*8 ..
        float4 a0 = xr0[0], a1 = xr0[1];
        float4 c0 = xr1[0], c1 = xr1[1];
        A.u[0] = pk_bf16(a0.x, a0.y); A.u[1] = pk_bf16(a0.z, a0.w);
        A.u[2] = pk_bf16(a1.x, a1.y); A.u[3] = pk_bf16(a1.z, a1.w);
        A.u[4] = pk_bf16(c0.x, c0.y); A.u[5] = pk_bf16(c0.z, c0.w);
        A.u[6] = pk_bf16(c1.x, c1.y); A.u[7] = pk_bf16(c1.z, c1.w);
        // B: 32x16 bf16 fragment from AmixT (ISA 16-bit layout, same K order as A)
        const uint4* bq = (const uint4*)(bp + k0 + g * 8);
        B.q[0] = bq[0];
        B.q[1] = bq[2];
        acc = __builtin_amdgcn_wmma_f32_16x16x32_bf16(
                  false, A.v, false, B.v, (short)0, acc, false, false);
    }
    // D layout: VGPR v -> row v + g*8, column r
#pragma unroll
    for (int v = 0; v < 8; ++v) {
        int m = rowBase + v + g * 8;
        xa[(size_t)m * RANK + r] = f2bf(acc[v]);
    }
}

// ---------------------------------------------------------------------------
// Kernel 3: out = x @ W^T + bias + (xa @ Bmix) ; 128x128 tile, 8 waves,
// each wave 2(M) x 4(N) WMMA tiles. Adapter fused as one rank-padded WMMA
// per output tile in the epilogue.
// ---------------------------------------------------------------------------
__global__ __launch_bounds__(256) void poly_main(
    const float* __restrict__ x,              // [16384][4096]
    const float* __restrict__ weight,         // [4096][4096]
    const float* __restrict__ bias,           // [4096]
    const unsigned short* __restrict__ xa,    // [16384][16] bf16
    const unsigned short* __restrict__ bmixT, // [8][4096][16] bf16
    float* __restrict__ out)                  // [16384][4096]
{
    __shared__ __align__(16) unsigned short Xs[128 * 32];   // 8 KB
    __shared__ __align__(16) unsigned short Wsh[128 * 32];  // 8 KB

    int tid  = threadIdx.x;
    int wave = tid >> 5;
    int lane = tid & 31;
    int g = lane >> 4;
    int r = lane & 15;
    int wm = wave >> 1;          // 0..3 -> 32-row slab
    int wn = wave & 1;           // 0..1 -> 64-col slab
    int m0 = blockIdx.x * 128;
    int n0 = blockIdx.y * 128;

    v8f acc[2][4];
    v8f vzero = {0.f, 0.f, 0.f, 0.f, 0.f, 0.f, 0.f, 0.f};
#pragma unroll
    for (int i = 0; i < 2; ++i)
#pragma unroll
        for (int j = 0; j < 4; ++j) acc[i][j] = vzero;

    // loader geometry: 1024 float4 per 128x32 f32 tile, 4 per thread
    int lrow[4], lcol[4];
#pragma unroll
    for (int i = 0; i < 4; ++i) {
        int ff = tid + i * 256;
        lrow[i] = ff >> 3;
        lcol[i] = ff & 7;
    }

    // --- preload first K-step into registers ---
    float4 gx[4], gw[4];
#pragma unroll
    for (int i = 0; i < 4; ++i) {
        gx[i] = *(const float4*)(x      + (size_t)(m0 + lrow[i]) * D_IN + lcol[i] * 4);
        gw[i] = *(const float4*)(weight + (size_t)(n0 + lrow[i]) * D_IN + lcol[i] * 4);
    }

    for (int k0 = 0; k0 < D_IN; k0 += 32) {
        __syncthreads();   // previous iteration's LDS reads complete
        // convert f32 -> bf16 in registers, stage to LDS (row stride 64 B)
#pragma unroll
        for (int i = 0; i < 4; ++i) {
            uint2 px = make_uint2(pk_bf16(gx[i].x, gx[i].y), pk_bf16(gx[i].z, gx[i].w));
            uint2 pw = make_uint2(pk_bf16(gw[i].x, gw[i].y), pk_bf16(gw[i].z, gw[i].w));
            *(uint2*)((char*)Xs  + lrow[i] * 64 + lcol[i] * 8) = px;
            *(uint2*)((char*)Wsh + lrow[i] * 64 + lcol[i] * 8) = pw;
        }
        __syncthreads();
        // register prefetch of next K-step overlaps the WMMAs below,
        // plus a cacheline prefetch two K-steps ahead (global_prefetch_b8)
        int k1 = k0 + 32;
        if (k1 < D_IN) {
#pragma unroll
            for (int i = 0; i < 4; ++i) {
                gx[i] = *(const float4*)(x      + (size_t)(m0 + lrow[i]) * D_IN + k1 + lcol[i] * 4);
                gw[i] = *(const float4*)(weight + (size_t)(n0 + lrow[i]) * D_IN + k1 + lcol[i] * 4);
            }
            if (k1 + 32 < D_IN) {
                __builtin_prefetch(x      + (size_t)(m0 + lrow[0]) * D_IN + k1 + 32 + lcol[0] * 4, 0, 1);
                __builtin_prefetch(weight + (size_t)(n0 + lrow[0]) * D_IN + k1 + 32 + lcol[0] * 4, 0, 1);
            }
        }
        // fragment loads (ds_load_b128 x2 each, ISA 16-bit layouts)
        FragB Af[2], Bf[4];
#pragma unroll
        for (int tm = 0; tm < 2; ++tm) {
            int row = wm * 32 + tm * 16 + r;
            Af[tm].q[0] = *(const uint4*)((char*)Xs + row * 64 + g * 16);
            Af[tm].q[1] = *(const uint4*)((char*)Xs + row * 64 + 32 + g * 16);
        }
#pragma unroll
        for (int tn = 0; tn < 4; ++tn) {
            int nrow = wn * 64 + tn * 16 + r;
            Bf[tn].q[0] = *(const uint4*)((char*)Wsh + nrow * 64 + g * 16);
            Bf[tn].q[1] = *(const uint4*)((char*)Wsh + nrow * 64 + 32 + g * 16);
        }
#pragma unroll
        for (int tm = 0; tm < 2; ++tm)
#pragma unroll
            for (int tn = 0; tn < 4; ++tn)
                acc[tm][tn] = __builtin_amdgcn_wmma_f32_16x16x32_bf16(
                    false, Af[tm].v, false, Bf[tn].v, (short)0, acc[tm][tn], false, false);
    }

    // ------------------- epilogue: adapter WMMA + bias + store ------------
    int batch = m0 >> 11;
    uint4 qz = make_uint4(0u, 0u, 0u, 0u);
#pragma unroll
    for (int tm = 0; tm < 2; ++tm) {
        int arow = m0 + wm * 32 + tm * 16 + r;
        FragB Ax;
        // rank 16 zero-padded to K=32: group g holds ranks g*8..g*8+7, upper half 0
        Ax.q[0] = *(const uint4*)(xa + (size_t)arow * RANK + g * 8);
        Ax.q[1] = qz;
#pragma unroll
        for (int tn = 0; tn < 4; ++tn) {
            int col = n0 + wn * 64 + tn * 16 + r;
            FragB Bx;
            {
                const uint4* bq = (const uint4*)(bmixT + (((size_t)batch << 12) + col) * RANK);
                Bx.q[0] = bq[g];
                Bx.q[1] = qz;
            }
            acc[tm][tn] = __builtin_amdgcn_wmma_f32_16x16x32_bf16(
                false, Ax.v, false, Bx.v, (short)0, acc[tm][tn], false, false);

            float bv = bias[col];
#pragma unroll
            for (int v = 0; v < 8; ++v) {
                int mrow = m0 + wm * 32 + tm * 16 + v + g * 8;
                out[(size_t)mrow * D_OUT + col] = acc[tm][tn][v] + bv;
            }
        }
    }
}

// ---------------------------------------------------------------------------
extern "C" void kernel_launch(void* const* d_in, const int* in_sizes, int n_in,
                              void* d_out, int out_size, void* d_ws, size_t ws_size,
                              hipStream_t stream) {
    (void)in_sizes; (void)n_in; (void)out_size; (void)ws_size;
    const float* x             = (const float*)d_in[0];
    const int*   task_ids      = (const int*)d_in[1];
    const float* weight        = (const float*)d_in[2];
    const float* bias          = (const float*)d_in[3];
    const float* module_logits = (const float*)d_in[4];
    const float* lora_a        = (const float*)d_in[5];
    const float* lora_b        = (const float*)d_in[6];
    float* out = (float*)d_out;

    char* ws = (char*)d_ws;
    unsigned short* amixT = (unsigned short*)(ws);                 // 1 MB  [8][16][4096]
    unsigned short* bmixT = (unsigned short*)(ws + (1u << 20));    // 1 MB  [8][4096][16]
    unsigned short* xabuf = (unsigned short*)(ws + (2u << 20));    // 512KB [16384][16]

    poly_mix_a<<<BS * D_IN / 256, 256, 0, stream>>>(lora_a, module_logits, task_ids, amixT);
    poly_mix_b<<<BS * D_OUT / 256, 256, 0, stream>>>(lora_b, module_logits, task_ids, bmixT);
    poly_xa<<<M_TOTAL / 128, 256, 0, stream>>>(x, amixT, xabuf);

    dim3 grid(M_TOTAL / 128, D_OUT / 128);
    poly_main<<<grid, 256, 0, stream>>>(x, weight, bias, xabuf, bmixT, out);
}
